// DetailedLSTMSentiment_90881507983763
// MI455X (gfx1250) — matrix operand order repaired and checked by reading, not compiled, not run on hardware
//
#include <hip/hip_runtime.h>
#include <hip/hip_bf16.h>
#include <stdint.h>

// ---------------------------------------------------------------------------
// DetailedLSTMSentiment on gfx1250 (MI455X), wave32 + WMMA bf16.
//   VOCAB=50257, EMB=512, HID=1024, NCLS=2, B=64, S=512
// Pipeline:
//   k_zero        : zero h double-buffer + grid-barrier counter (each call)
//   k_f32_to_bf16 : convert w_ih / w_hh to bf16 workspace copies
//   k_gather_embed: fused embedding gather + f32->bf16, layout [s*64+b][512]
//   k_gemm_xg     : xg[s][b][4H] = embedded @ w_ih^T + b_ih   (WMMA bf16)
//   k_lstm_scan   : persistent kernel, 512 steps, w_hh slice in LDS,
//                   c-state in registers, grid barrier per step (WMMA bf16)
//   k_classifier  : out = h_T @ cls_w^T + cls_b
// ---------------------------------------------------------------------------

typedef __attribute__((ext_vector_type(16))) __bf16 v16bf;
typedef __attribute__((ext_vector_type(8)))  float  v8f;

#define BATCH 64
#define SEQ   512
#define EMBD  512
#define HID   1024
#define G4    4096
#define NBLK_SCAN 32

union ABu { v16bf v; uint4 q[2]; };

__device__ __forceinline__ unsigned short f2bf(float f) {
  unsigned u  = __builtin_bit_cast(unsigned, f);
  unsigned rb = 0x7FFFu + ((u >> 16) & 1u);   // round-to-nearest-even
  return (unsigned short)((u + rb) >> 16);
}
__device__ __forceinline__ float bf2f(unsigned short h) {
  return __builtin_bit_cast(float, (unsigned)h << 16);
}
__device__ __forceinline__ float sigmoidf_(float x) {
  return 1.0f / (1.0f + __expf(-x));
}
__device__ __forceinline__ uint4 pack8(float4 f0, float4 f1) {
  uint4 o;
  o.x = (unsigned)f2bf(f0.x) | ((unsigned)f2bf(f0.y) << 16);
  o.y = (unsigned)f2bf(f0.z) | ((unsigned)f2bf(f0.w) << 16);
  o.z = (unsigned)f2bf(f1.x) | ((unsigned)f2bf(f1.y) << 16);
  o.w = (unsigned)f2bf(f1.z) | ((unsigned)f2bf(f1.w) << 16);
  return o;
}

// --------------------------------------------------------------------------
__global__ void k_zero_u32(unsigned* p, int n) {
  int i = blockIdx.x * blockDim.x + threadIdx.x;
  if (i < n) p[i] = 0u;
}

__global__ void k_f32_to_bf16(const float* __restrict__ src,
                              unsigned short* __restrict__ dst, int nchunks) {
  int stride = gridDim.x * blockDim.x;
  for (int c = blockIdx.x * blockDim.x + threadIdx.x; c < nchunks; c += stride) {
    float4 f0 = ((const float4*)src)[(size_t)c * 2 + 0];
    float4 f1 = ((const float4*)src)[(size_t)c * 2 + 1];
    ((uint4*)dst)[c] = pack8(f0, f1);
  }
}

// embBf[r][e] = bf16(emb[x[b][s]][e]),  r = s*64 + b  (rows = scan-time order)
__global__ void k_gather_embed(const int* __restrict__ x,
                               const float* __restrict__ emb,
                               unsigned short* __restrict__ embBf) {
  const int nch = (BATCH * SEQ * EMBD) / 8;
  int stride = gridDim.x * blockDim.x;
  for (int c = blockIdx.x * blockDim.x + threadIdx.x; c < nch; c += stride) {
    int r  = c >> 6;                // 64 chunks of 8 per 512-wide row
    int e0 = (c & 63) << 3;
    int b  = r & (BATCH - 1);
    int s  = r >> 6;
    int tok = x[b * SEQ + s];
    const float4* ep = (const float4*)(emb + (size_t)tok * EMBD + e0);
    ((uint4*)embBf)[c] = pack8(ep[0], ep[1]);
  }
}

// --------------------------------------------------------------------------
// xg[r][n] = sum_e embBf[r][e] * w_ih[n][e] + b_ih[n]   (r = s*64+b, n in 4H)
// 64x64 macro-tile per block; 8 waves; wave = (mtile, 2 ntiles); K=512.
__global__ __launch_bounds__(256) void k_gemm_xg(
    const unsigned short* __restrict__ embBf,
    const unsigned short* __restrict__ wihBf,
    const float* __restrict__ bih,
    float* __restrict__ xg) {
  const int lane = threadIdx.x & 31;
  const int w    = threadIdx.x >> 5;
  const int mt   = w & 3;
  const int ngrp = w >> 2;                       // 0..1
  const int m0   = blockIdx.y * 64 + mt * 16;
  const int arow   = m0 + (lane & 15);
  const int achunk = (lane >> 4) * 8;            // A: ISA 16-bit A layout
  const int bchunk = (lane >> 4) * 8;            // B: ISA 16-bit B layout
  const unsigned short* ap = embBf + (size_t)arow * EMBD + achunk;

  for (int nn = 0; nn < 2; ++nn) {
    const int n0   = blockIdx.x * 64 + (ngrp * 2 + nn) * 16;
    const int ncol = n0 + (lane & 15);
    const unsigned short* bp = wihBf + (size_t)ncol * EMBD + bchunk;

    v8f acc;
    float bb = bih[ncol];
#pragma unroll
    for (int r = 0; r < 8; ++r) acc[r] = bb;

#pragma unroll
    for (int kk = 0; kk < EMBD / 32; ++kk) {
      ABu a, b;
      a.q[0] = *(const uint4*)(ap + kk * 32);
      a.q[1] = *(const uint4*)(ap + kk * 32 + 16);
      b.q[0] = *(const uint4*)(bp + kk * 32);
      b.q[1] = *(const uint4*)(bp + kk * 32 + 16);
      acc = __builtin_amdgcn_wmma_f32_16x16x32_bf16(
          false, a.v, false, b.v, (short)0, acc, false, false);
    }
    const int mrow = m0 + (lane >> 4) * 8;
#pragma unroll
    for (int r = 0; r < 8; ++r)
      xg[(size_t)(mrow + r) * G4 + ncol] = acc[r];
  }
}

// --------------------------------------------------------------------------
// Persistent LSTM scan. 32 blocks x 256 threads (8 waves).
// Block bk owns hidden columns [bk*32, bk*32+32). Its w_hh slice
// (4 gates x 32 rows x 1024 k, bf16) is staged in LDS once (≈258 KB of the
// 320 KB WGP LDS) and reused across all 512 steps.
// Wave (mt, jt) computes 16x16 tiles of all 4 gates for batch rows mt*16..
// and hidden cols jt*16.., keeps cell state c in registers.
__global__ __launch_bounds__(256) void k_lstm_scan(
    const float* __restrict__ xg,
    const unsigned short* __restrict__ whhBf,
    const float* __restrict__ bhh,
    unsigned short* __restrict__ hbuf,       // double buffer [2][64][1024] bf16
    unsigned* __restrict__ bar) {
  extern __shared__ unsigned short sW[];     // [128][HID+8] bf16 (+8 pad -> no bank clash)
  const int LROW = HID + 8;
  const int j0 = blockIdx.x * 32;

  // Stage w_hh slice: LDS row (g*32 + jl) <- w_hh[g*1024 + j0 + jl][:]
  for (int t = threadIdx.x; t < 128 * (HID / 8); t += blockDim.x) {
    int rrow = t >> 7;
    int qc   = t & 127;
    int ngl  = (rrow >> 5) * HID + j0 + (rrow & 31);
    *(uint4*)(sW + (size_t)rrow * LROW + qc * 8) =
        *(const uint4*)(whhBf + (size_t)ngl * HID + qc * 8);
  }
  __syncthreads();

  const int lane = threadIdx.x & 31;
  const int w    = threadIdx.x >> 5;
  const int mt   = w & 3;
  const int jt   = (w >> 2) & 1;
  const int m0   = mt * 16;
  const int jl0  = jt * 16;
  const int nl   = lane & 15;
  const int hi   = lane >> 4;
  const int arow   = m0 + nl;
  const int achunk = hi * 8;
  const int bchunk = hi * 8;
  const int mbase  = m0 + hi * 8;
  const int jcol   = j0 + jl0 + nl;

  float bb[4];
#pragma unroll
  for (int g = 0; g < 4; ++g) bb[g] = bhh[g * HID + jcol];

  v8f cstate;
#pragma unroll
  for (int r = 0; r < 8; ++r) cstate[r] = 0.0f;

  for (int s = 0; s < SEQ; ++s) {
    const float* xgs = xg + (size_t)s * BATCH * G4;
    const unsigned short* hsrc = hbuf + (size_t)(s & 1) * BATCH * HID;
    unsigned short*       hdst = hbuf + (size_t)((s + 1) & 1) * BATCH * HID;

    // accumulator init: xg (has b_ih) + b_hh
    v8f acc[4];
#pragma unroll
    for (int g = 0; g < 4; ++g) {
      int ncol = g * HID + jcol;
#pragma unroll
      for (int r = 0; r < 8; ++r)
        acc[g][r] = xgs[(size_t)(mbase + r) * G4 + ncol] + bb[g];
    }

    // gates += h @ w_hh^T : K = 1024 = 32 x (16x16x32 bf16 WMMA), x4 gates
    const unsigned short* ap0 = hsrc + (size_t)arow * HID + achunk;
#pragma unroll 4
    for (int kk = 0; kk < HID / 32; ++kk) {
      ABu a;
      a.q[0] = *(const uint4*)(ap0 + kk * 32);
      a.q[1] = *(const uint4*)(ap0 + kk * 32 + 16);
#pragma unroll
      for (int g = 0; g < 4; ++g) {
        ABu bm;
        const unsigned short* bp =
            sW + (size_t)(g * 32 + jl0 + nl) * LROW + kk * 32 + bchunk;
        bm.q[0] = *(const uint4*)(bp);
        bm.q[1] = *(const uint4*)(bp + 16);
        acc[g] = __builtin_amdgcn_wmma_f32_16x16x32_bf16(
            false, a.v, false, bm.v, (short)0, acc[g], false, false);
      }
    }

    // elementwise cell update (gate order f,i,g,o as in reference)
#pragma unroll
    for (int r = 0; r < 8; ++r) {
      float fg = sigmoidf_(acc[0][r]);
      float ig = sigmoidf_(acc[1][r]);
      float gg = tanhf(acc[2][r]);
      float og = sigmoidf_(acc[3][r]);
      float c  = fg * cstate[r] + ig * gg;
      cstate[r] = c;
      float h = og * tanhf(c);
      hdst[(size_t)(mbase + r) * HID + jcol] = f2bf(h);
    }

    if (s + 1 < SEQ)  // warm L2/WGP$ for next step's xg slice
      __builtin_prefetch(xgs + (size_t)BATCH * G4 + (size_t)mbase * G4 + jcol, 0, 1);

    // ---- device-wide barrier (monotonic counter, zeroed per launch) ----
    __threadfence();
    __syncthreads();
    if (threadIdx.x == 0) {
      atomicAdd(bar, 1u);
      unsigned target = (unsigned)(s + 1) * NBLK_SCAN;
      while (__hip_atomic_load(bar, __ATOMIC_ACQUIRE,
                               __HIP_MEMORY_SCOPE_AGENT) < target) {
        __builtin_amdgcn_s_sleep(1);
      }
    }
    __syncthreads();
    __threadfence();
  }
}

// --------------------------------------------------------------------------
__global__ void k_classifier(const unsigned short* __restrict__ hfin,
                             const float* __restrict__ clsw,
                             const float* __restrict__ clsb,
                             float* __restrict__ out) {
  int t = threadIdx.x;
  if (t >= BATCH * 2) return;
  int b = t >> 1, c = t & 1;
  float acc = clsb[c];
  for (int k = 0; k < HID; ++k)
    acc += bf2f(hfin[b * HID + k]) * clsw[c * HID + k];
  out[b * 2 + c] = acc;
}

// --------------------------------------------------------------------------
extern "C" void kernel_launch(void* const* d_in, const int* in_sizes, int n_in,
                              void* d_out, int out_size, void* d_ws, size_t ws_size,
                              hipStream_t stream) {
  (void)in_sizes; (void)n_in; (void)out_size; (void)ws_size;
  const int*   x    = (const int*)  d_in[0];
  const float* emb  = (const float*)d_in[1];
  const float* wih  = (const float*)d_in[2];
  const float* bih  = (const float*)d_in[3];
  const float* whh  = (const float*)d_in[4];
  const float* bhh  = (const float*)d_in[5];
  const float* clsw = (const float*)d_in[6];
  const float* clsb = (const float*)d_in[7];
  float* out = (float*)d_out;

  // workspace layout (bytes)
  char* ws = (char*)d_ws;
  unsigned short* embBf = (unsigned short*)(ws);                 // 32 MB  [32768][512]
  unsigned short* wihBf = (unsigned short*)(ws + 33554432ull);   //  4 MB  [4096][512]
  unsigned short* whhBf = (unsigned short*)(ws + 37748736ull);   //  8 MB  [4096][1024]
  unsigned short* hbuf  = (unsigned short*)(ws + 46137344ull);   // 256 KB [2][64][1024]
  unsigned*       bar   = (unsigned*)     (ws + 46399488ull);    // 256 B
  float*          xg    = (float*)        (ws + 46399744ull);    // 512 MB [512][64][4096]

  // zero h double-buffer + barrier counter every call (deterministic replay)
  k_zero_u32<<<(65600 + 255) / 256, 256, 0, stream>>>(
      (unsigned*)(ws + 46137344ull), 65600);
  k_f32_to_bf16<<<1024, 256, 0, stream>>>(wih, wihBf, (G4 * EMBD) / 8);
  k_f32_to_bf16<<<1024, 256, 0, stream>>>(whh, whhBf, (G4 * HID) / 8);
  k_gather_embed<<<4096, 256, 0, stream>>>(x, emb, embBf);

  dim3 g1(G4 / 64, (BATCH * SEQ) / 64);          // 64 x 512 blocks
  k_gemm_xg<<<g1, 256, 0, stream>>>(embBf, wihBf, bih, xg);

  size_t ldsBytes = 128ull * (HID + 8) * sizeof(unsigned short);  // 264,192 B
  k_lstm_scan<<<NBLK_SCAN, 256, ldsBytes, stream>>>(xg, whhBf, bhh, hbuf, bar);

  // final h lives in buffer parity (SEQ)&1 == 0
  k_classifier<<<1, 128, 0, stream>>>(hbuf, clsw, clsb, out);
}
